// QuantizedLinear_51943334478221
// MI455X (gfx1250) — 8-bit matrix operand order repaired and checked by reading, compiled, not run on hardware
//
#include <hip/hip_runtime.h>
#include <stdint.h>


#define MDIM 16384
#define KDIM 4096
#define NDIM 4096

typedef __attribute__((ext_vector_type(8))) int   v8i;
typedef __attribute__((ext_vector_type(4))) int   v4i;
typedef __attribute__((ext_vector_type(2))) int   v2i;
typedef __attribute__((ext_vector_type(4))) float v4f;

#ifndef __has_builtin
#define __has_builtin(x) 0
#endif

#if __has_builtin(__builtin_amdgcn_global_load_async_to_lds_b128)
#define HAVE_ASYNC 1
#else
#define HAVE_ASYNC 0
#endif

typedef __attribute__((address_space(1))) v4i gv4i;
typedef __attribute__((address_space(3))) v4i lv4i;

__device__ __forceinline__ void cp16_g2lds(void* lds_dst, const void* g_src) {
#if HAVE_ASYNC
  __builtin_amdgcn_global_load_async_to_lds_b128((gv4i*)g_src, (lv4i*)lds_dst, 0, 0);
#else
  *(v4i*)lds_dst = *(const v4i*)g_src;
#endif
}

__device__ __forceinline__ void wait_async_all() {
#if HAVE_ASYNC
#if __has_builtin(__builtin_amdgcn_s_wait_asynccnt)
  __builtin_amdgcn_s_wait_asynccnt(0);
#else
  asm volatile("s_wait_asynccnt 0" ::: "memory");
#endif
#endif
}

__global__ __launch_bounds__(256) void quant_x_kernel(const float* __restrict__ x,
                                                      int8_t* __restrict__ xq,
                                                      float* __restrict__ xs) {
  __shared__ float red[256];
  const int token = blockIdx.x;
  const int tid   = threadIdx.x;
  const float* xrow = x + (size_t)token * KDIM;

  v4f v[4];
  float m = 0.0f;
#pragma unroll
  for (int i = 0; i < 4; ++i) {
    v[i] = *(const v4f*)(xrow + tid * 16 + i * 4);
#pragma unroll
    for (int j = 0; j < 4; ++j) m = fmaxf(m, fabsf(v[i][j]));
  }
  red[tid] = m;
  __syncthreads();
#pragma unroll
  for (int s = 128; s > 0; s >>= 1) {
    if (tid < s) red[tid] = fmaxf(red[tid], red[tid + s]);
    __syncthreads();
  }
  const float maxv = fmaxf(red[0], 1e-20f);
  const float inv  = 127.0f / maxv;
  if (tid == 0) xs[token] = maxv * (1.0f / 127.0f);

  uint32_t* qout = (uint32_t*)(xq + (size_t)token * KDIM) + tid * 4;
#pragma unroll
  for (int i = 0; i < 4; ++i) {
    uint32_t p = 0;
#pragma unroll
    for (int j = 0; j < 4; ++j) {
      float q = rintf(v[i][j] * inv);
      q = fminf(fmaxf(q, -128.0f), 127.0f);
      p |= ((uint32_t)(((int)q) & 0xFF)) << (8 * j);
    }
    qout[i] = p;
  }
}

__global__ __launch_bounds__(256) void pack_w_kernel(const int* __restrict__ w,
                                                     uint32_t* __restrict__ wq,
                                                     int n4) {
  const int i = blockIdx.x * 256 + threadIdx.x;
  if (i >= n4) return;
  v4i c = *(const v4i*)(w + (size_t)i * 4);
  wq[i] = (uint32_t)(c[0] & 0xFF)        | ((uint32_t)(c[1] & 0xFF) << 8) |
          ((uint32_t)(c[2] & 0xFF) << 16) | ((uint32_t)(c[3] & 0xFF) << 24);
}

__global__ __launch_bounds__(256) void gemm_iu8_kernel(
    const int8_t* __restrict__ xq, const int8_t* __restrict__ wq,
    const float* __restrict__ xs, const float* __restrict__ sw,
    const float* __restrict__ bias, float* __restrict__ out) {
  __shared__ __align__(16) int8_t As[2][128 * 64];
  __shared__ __align__(16) int8_t Bs[2][128 * 64];

  const int tid  = threadIdx.x;
  const int wave = tid >> 5;
  const int lane = tid & 31;
  const int wm   = wave >> 1;
  const int wn   = wave & 1;
  const int r    = lane & 15;
  const int h    = lane >> 4;

  const int bm = blockIdx.y * 128;
  const int bn = blockIdx.x * 128;

  const int lrow = tid >> 2;
  const int lcol = (tid & 3) * 16;
  const int8_t* agp = xq + (size_t)(bm + lrow) * KDIM + lcol;
  const int8_t* bgp = wq + (size_t)(bn + lrow) * KDIM + lcol;

  v8i acc[2][4] = {};

  {
    cp16_g2lds(&As[0][lrow * 64 + lcol],        agp);
    cp16_g2lds(&As[0][(lrow + 64) * 64 + lcol], agp + (size_t)64 * KDIM);
    cp16_g2lds(&Bs[0][lrow * 64 + lcol],        bgp);
    cp16_g2lds(&Bs[0][(lrow + 64) * 64 + lcol], bgp + (size_t)64 * KDIM);
  }
  wait_async_all();
  __syncthreads();

  const int KSTEPS = KDIM / 64;
#pragma unroll 1
  for (int kk = 0; kk < KSTEPS; ++kk) {
    const int cur = kk & 1;
    if (kk + 1 < KSTEPS) {
      const int nxt = cur ^ 1;
      const size_t ko = (size_t)(kk + 1) * 64;
      cp16_g2lds(&As[nxt][lrow * 64 + lcol],        agp + ko);
      cp16_g2lds(&As[nxt][(lrow + 64) * 64 + lcol], agp + (size_t)64 * KDIM + ko);
      cp16_g2lds(&Bs[nxt][lrow * 64 + lcol],        bgp + ko);
      cp16_g2lds(&Bs[nxt][(lrow + 64) * 64 + lcol], bgp + (size_t)64 * KDIM + ko);
    }

    const int8_t* a_base = &As[cur][0];
    const int8_t* b_base = &Bs[cur][0];

    v8i afrag[2];
#pragma unroll
    for (int i = 0; i < 2; ++i) {
      const int arow = wm * 32 + i * 16 + r;
#pragma unroll
      for (int j = 0; j < 4; ++j) {
        v2i d = *(const v2i*)(a_base + arow * 64 + h * 16 + j * 8 + (j >> 1) * 16);
        afrag[i][2 * j]     = d[0];
        afrag[i][2 * j + 1] = d[1];
      }
    }

#pragma unroll
    for (int n = 0; n < 4; ++n) {
      const int brow = wn * 64 + n * 16 + r;
      v4i d0 = *(const v4i*)(b_base + brow * 64 + h * 16);
      v4i d1 = *(const v4i*)(b_base + brow * 64 + 32 + h * 16);
      v8i bfrag;
      bfrag[0] = d0[0]; bfrag[1] = d0[1]; bfrag[2] = d0[2]; bfrag[3] = d0[3];
      bfrag[4] = d1[0]; bfrag[5] = d1[1]; bfrag[6] = d1[2]; bfrag[7] = d1[3];
#pragma unroll
      for (int i = 0; i < 2; ++i) {
        acc[i][n] = __builtin_amdgcn_wmma_i32_16x16x64_iu8(
             true, afrag[i],  true, bfrag,
            acc[i][n],  false,  false);
      }
    }

    wait_async_all();
    __syncthreads();
  }

  const int m0 = bm + wm * 32;
  const int n0 = bn + wn * 64;

  float xsv[2][8];
#pragma unroll
  for (int i = 0; i < 2; ++i)
#pragma unroll
    for (int j = 0; j < 8; ++j)
      xsv[i][j] = xs[m0 + i * 16 + h * 8 + j];

#pragma unroll
  for (int n = 0; n < 4; ++n) {
    const int ncol = n0 + n * 16 + r;
    const float swv = sw[ncol];
    const float bv  = bias[ncol];
#pragma unroll
    for (int i = 0; i < 2; ++i) {
      const int mrow = m0 + i * 16 + h * 8;
#pragma unroll
      for (int j = 0; j < 8; ++j) {
        float o = (float)acc[i][n][j] * xsv[i][j] * swv + bv;
        out[(size_t)(mrow + j) * NDIM + ncol] = o;
      }
    }
  }
}

extern "C" void kernel_launch(void* const* d_in, const int* in_sizes, int n_in,
                              void* d_out, int out_size, void* d_ws, size_t ws_size,
                              hipStream_t stream) {
  (void)in_sizes; (void)n_in; (void)out_size; (void)ws_size;

  const float* x    = (const float*)d_in[0];
  const int*   w    = (const int*)d_in[1];
  const float* sw   = (const float*)d_in[2];
  const float* bias = (const float*)d_in[3];
  float* out = (float*)d_out;

  int8_t* xq = (int8_t*)d_ws;
  int8_t* wq = xq + (size_t)MDIM * KDIM;
  float*  xs = (float*)(wq + (size_t)NDIM * KDIM);

  quant_x_kernel<<<MDIM, 256, 0, stream>>>(x, xq, xs);

  const int n4 = (NDIM * KDIM) / 4;
  pack_w_kernel<<<n4 / 256, 256, 0, stream>>>(w, (uint32_t*)wq, n4);

  dim3 grid(NDIM / 128, MDIM / 128);
  gemm_iu8_kernel<<<grid, 256, 0, stream>>>(xq, wq, xs, sw, bias, out);
}
